// QAG_38388417692074
// MI455X (gfx1250) — matrix operand order repaired and checked by reading, not compiled, not run on hardware
//
#include <hip/hip_runtime.h>
#include <hip/hip_bf16.h>
#include <cstdint>
#include <cstddef>

// ---------------- problem geometry (matches reference) ----------------
#define GN    8
#define NPG   2048
#define DD    512
#define HH    4          // heads
#define CHD   128        // head dim
#define TT    8          // graph tokens
#define E2C   262144     // edges stage 2
#define E1C   (GN * NPG)         // 16384  (question -> node star)
#define E3C   (GN * TT * NPG)    // 131072 (node -> token)
#define N1C   (GN * (NPG + 1))   // 16392
#define N2C   (GN * NPG)         // 16384
#define N3C   (GN * (NPG + TT))  // 16448
#define NMAXC N3C

typedef __bf16 bf16_t;
typedef __attribute__((ext_vector_type(16))) __bf16 bf16x16;
typedef __attribute__((ext_vector_type(8)))  float  f32x8;

// gfx1250 async global->LDS path (ASYNCcnt-tracked), guarded so a missing
// builtin falls back to register-staged LDS stores.
#if __has_builtin(__builtin_amdgcn_global_load_async_to_lds_b128)
#define HAVE_ASYNC_LDS 1
// exact parameter types per hipcc diagnostic: vector_size(16) int, AS(1)/AS(3)
typedef int i32x4v __attribute__((vector_size(16)));
typedef __attribute__((address_space(1))) i32x4v as1_i32x4;
typedef __attribute__((address_space(3))) i32x4v as3_i32x4;
#else
#define HAVE_ASYNC_LDS 0
#endif

#if __has_builtin(__builtin_amdgcn_s_wait_asynccnt)
#define WAIT_ASYNCCNT0() __builtin_amdgcn_s_wait_asynccnt(0)
#else
#define WAIT_ASYNCCNT0() asm volatile("s_wait_asynccnt 0x0" ::: "memory")
#endif

// =====================================================================
// WMMA GEMM:  C[M,N] = A[M,K] @ B[K,N] (+bias) ; A bf16 row-major,
// B supplied pre-TRANSPOSED bf16 (Bt[N,K]) so both LDS tiles are
// K-contiguous and fragment loads are ds_load_b128s.
// Block tile 128x128, K staged in 32-wide slices with ping-pong LDS
// double buffering (40KB) so stage s+1 tile movement overlaps the
// 8 v_wmma_f32_16x16x32_bf16 of stage s. 256 threads = 8 waves (wave32),
// wave tile 32x64 = 8 independent accumulators.
// =====================================================================
#define BKS 32

template <bool RELU, typename OutT>
__global__ __launch_bounds__(256) void gemm_wmma(
    const bf16_t* __restrict__ A, const bf16_t* __restrict__ Bt,
    const float* __restrict__ bias, OutT* __restrict__ C,
    int M, int N, int K) {
  __shared__ bf16_t As[2][128][BKS + 8];  // +8 pad: rows stay 16B aligned,
  __shared__ bf16_t Bs[2][128][BKS + 8];  // conflict-free b128 fragment loads

  const int tid  = threadIdx.x;
  const int lane = tid & 31;
  const int wave = tid >> 5;
  const int wm   = wave & 3;   // 4 wave-rows of 32 M
  const int wn   = wave >> 2;  // 2 wave-cols of 64 N
  const int half = lane >> 4;  // 0/1 (K-half selector per ISA layout)
  const int l16  = lane & 15;

  const int tileM = blockIdx.x * 128;
  const int tileN = blockIdx.y * 128;
  const int S     = K / BKS;  // K assumed multiple of 32 (512 here)

  // per-thread tile-load coordinates: 2 x 16B chunks of A + 2 of B per stage
  int rA[2], cA[2];
#pragma unroll
  for (int i = 0; i < 2; ++i) {
    int c = tid + i * 256;       // 0..511
    rA[i] = c >> 2;              // row 0..127
    cA[i] = (c & 3) * 8;         // col offset in bf16 elems
  }

  f32x8 acc[2][4];
#pragma unroll
  for (int i = 0; i < 2; ++i)
#pragma unroll
    for (int j = 0; j < 4; ++j) acc[i][j] = (f32x8){0, 0, 0, 0, 0, 0, 0, 0};

#if HAVE_ASYNC_LDS
  // async path: rows are clamped instead of zero-filled; clamped rows only
  // feed output rows >= M / >= N which are never stored.
  auto issue_async = [&](int s, int buf) {
    int k0 = s * BKS;
#pragma unroll
    for (int i = 0; i < 2; ++i) {
      int gm = tileM + rA[i]; if (gm >= M) gm = M - 1;
      int gn = tileN + rA[i]; if (gn >= N) gn = N - 1;
      __builtin_amdgcn_global_load_async_to_lds_b128(
          (as1_i32x4*)(A + (size_t)gm * K + k0 + cA[i]),
          (as3_i32x4*)&As[buf][rA[i]][cA[i]], 0, 0);
      __builtin_amdgcn_global_load_async_to_lds_b128(
          (as1_i32x4*)(Bt + (size_t)gn * K + k0 + cA[i]),
          (as3_i32x4*)&Bs[buf][rA[i]][cA[i]], 0, 0);
    }
  };
  issue_async(0, 0);
  WAIT_ASYNCCNT0();
  __syncthreads();
#else
  uint4 stA[2], stB[2];
  auto fetch_regs = [&](int s) {
    int k0 = s * BKS;
#pragma unroll
    for (int i = 0; i < 2; ++i) {
      int gm = tileM + rA[i];
      stA[i] = (gm < M) ? *(const uint4*)(A + (size_t)gm * K + k0 + cA[i])
                        : (uint4){0, 0, 0, 0};
      int gn = tileN + rA[i];
      stB[i] = (gn < N) ? *(const uint4*)(Bt + (size_t)gn * K + k0 + cA[i])
                        : (uint4){0, 0, 0, 0};
    }
  };
  auto commit_lds = [&](int buf) {
#pragma unroll
    for (int i = 0; i < 2; ++i) {
      *(uint4*)&As[buf][rA[i]][cA[i]] = stA[i];
      *(uint4*)&Bs[buf][rA[i]][cA[i]] = stB[i];
    }
  };
  fetch_regs(0);
  commit_lds(0);
  __syncthreads();
#endif

  int cur = 0;
  for (int s = 0; s < S; ++s) {
    // ---- start stage s+1 tile movement before consuming stage s ----
#if HAVE_ASYNC_LDS
    if (s + 1 < S) issue_async(s + 1, cur ^ 1);
#else
    if (s + 1 < S) fetch_regs(s + 1);
#endif
    if (s + 2 < S) {  // pull stage s+2 toward L2 (global_prefetch_b8)
      int k2 = (s + 2) * BKS;
      int gm = tileM + rA[0]; if (gm >= M) gm = M - 1;
      int gn = tileN + rA[0]; if (gn >= N) gn = N - 1;
      __builtin_prefetch(A + (size_t)gm * K + k2 + cA[0], 0, 1);
      __builtin_prefetch(Bt + (size_t)gn * K + k2 + cA[0], 0, 1);
    }

    // ---- compute stage s: 8 WMMAs per wave ----
    bf16x16 af[2], bfr[4];
#pragma unroll
    for (int mi = 0; mi < 2; ++mi) {
      int r  = wm * 32 + mi * 16 + l16;
      int kb = half * 8;  // ISA 16-bit A layout: element halves hold K, K+16
      __builtin_memcpy(&af[mi], &As[cur][r][kb], 16);
      __builtin_memcpy((char*)&af[mi] + 16, &As[cur][r][kb + 16], 16);
    }
#pragma unroll
    for (int ni = 0; ni < 4; ++ni) {
      int r  = wn * 64 + ni * 16 + l16;
      int kb = half * 8;  // ISA 16-bit B layout: element halves hold K, K+16
      __builtin_memcpy(&bfr[ni], &Bs[cur][r][kb], 16);
      __builtin_memcpy((char*)&bfr[ni] + 16, &Bs[cur][r][kb + 16], 16);
    }
#pragma unroll
    for (int mi = 0; mi < 2; ++mi)
#pragma unroll
      for (int ni = 0; ni < 4; ++ni)
        acc[mi][ni] = __builtin_amdgcn_wmma_f32_16x16x32_bf16(
            false, af[mi], false, bfr[ni], (short)0, acc[mi][ni], false, false);

    // ---- make stage s+1 visible, flip buffers ----
#if HAVE_ASYNC_LDS
    if (s + 1 < S) WAIT_ASYNCCNT0();
#else
    if (s + 1 < S) commit_lds(cur ^ 1);
#endif
    __syncthreads();
    cur ^= 1;
  }

  // epilogue: C layout -> m = base + half*8 + r ; n = base + (lane&15)
#pragma unroll
  for (int mi = 0; mi < 2; ++mi) {
#pragma unroll
    for (int ni = 0; ni < 4; ++ni) {
      int n = tileN + wn * 64 + ni * 16 + l16;
      float bv = bias ? bias[n] : 0.0f;
#pragma unroll
      for (int r = 0; r < 8; ++r) {
        int m = tileM + wm * 32 + mi * 16 + half * 8 + r;
        if (m < M && n < N) {
          float v = acc[mi][ni][r] + bv;
          if (RELU) v = v > 0.0f ? v : 0.0f;
          C[(size_t)m * N + n] = (OutT)v;
        }
      }
    }
  }
}

// =====================================================================
// elementwise / utility kernels
// =====================================================================
__global__ __launch_bounds__(256) void cast_f32_bf16_kernel(
    const float* __restrict__ s, bf16_t* __restrict__ d, size_t n) {
  size_t i = (size_t)blockIdx.x * 256 + threadIdx.x;
  if (i < n) d[i] = (bf16_t)s[i];
}

__global__ __launch_bounds__(256) void transpose_cast_kernel(
    const float* __restrict__ W, bf16_t* __restrict__ Wt, int K, int N) {
  size_t i = (size_t)blockIdx.x * 256 + threadIdx.x;
  if (i >= (size_t)K * N) return;
  int k = (int)(i / N), n = (int)(i % N);
  Wt[(size_t)n * K + k] = (bf16_t)W[i];
}

__global__ __launch_bounds__(256) void fill_u32_kernel(unsigned* p, unsigned v, size_t n) {
  size_t i = (size_t)blockIdx.x * 256 + threadIdx.x;
  if (i < n) p[i] = v;
}

// monotonic float<->uint encode for atomicMax-based segment max
__device__ __forceinline__ unsigned ordEnc(float f) {
  unsigned u = __float_as_uint(f);
  return (u & 0x80000000u) ? ~u : (u | 0x80000000u);
}
__device__ __forceinline__ float ordDec(unsigned u) {
  unsigned b = (u & 0x80000000u) ? (u & 0x7fffffffu) : ~u;
  return __uint_as_float(b);
}

// =====================================================================
// attention edge kernels: one wave32 per (edge, head); head dim 128 =
// 32 lanes x float4
// =====================================================================
__global__ __launch_bounds__(256) void edge_alpha_kernel(
    const float* __restrict__ q, const float* __restrict__ k,
    const bf16_t* __restrict__ eproj, const int* __restrict__ src,
    const int* __restrict__ dst, float* __restrict__ alpha,
    unsigned* __restrict__ amax, int EH, float scale) {
  int pair = blockIdx.x * 8 + (threadIdx.x >> 5);
  if (pair >= EH) return;
  int lane = threadIdx.x & 31;
  int e = pair >> 2, h = pair & 3;
  int s = src[e], d = dst[e];
  size_t co = (size_t)h * CHD + lane * 4;
  float4 qv = *(const float4*)(q + (size_t)d * DD + co);
  float4 kv = *(const float4*)(k + (size_t)s * DD + co);
  if (eproj) {
    union { unsigned long long u; __bf16 hx[4]; } ev;
    ev.u = *(const unsigned long long*)(eproj + (size_t)e * DD + co);
    kv.x += (float)ev.hx[0]; kv.y += (float)ev.hx[1];
    kv.z += (float)ev.hx[2]; kv.w += (float)ev.hx[3];
  }
  float sum = qv.x * kv.x + qv.y * kv.y + qv.z * kv.z + qv.w * kv.w;
#pragma unroll
  for (int off = 16; off; off >>= 1) sum += __shfl_xor(sum, off, 32);
  if (lane == 0) {
    float a = sum * scale;
    alpha[pair] = a;
    atomicMax(amax + (size_t)d * HH + h, ordEnc(a));
  }
}

__global__ __launch_bounds__(256) void edge_exp_kernel(
    float* __restrict__ alpha, const unsigned* __restrict__ amax,
    float* __restrict__ den, const int* __restrict__ dst, int EH) {
  int i = blockIdx.x * 256 + threadIdx.x;
  if (i >= EH) return;
  int e = i >> 2, h = i & 3;
  int d = dst[e];
  float ex = __expf(alpha[i] - ordDec(amax[(size_t)d * HH + h]));
  alpha[i] = ex;
  atomicAdd(den + (size_t)d * HH + h, ex);
}

__global__ __launch_bounds__(256) void edge_scatter_kernel(
    const float* __restrict__ alpha, const float* __restrict__ den,
    const float* __restrict__ v, const bf16_t* __restrict__ eproj,
    const int* __restrict__ src, const int* __restrict__ dst,
    float* __restrict__ out, int EH) {
  int pair = blockIdx.x * 8 + (threadIdx.x >> 5);
  if (pair >= EH) return;
  int lane = threadIdx.x & 31;
  int e = pair >> 2, h = pair & 3;
  int s = src[e], d = dst[e];
  float w = alpha[pair] / (den[(size_t)d * HH + h] + 1e-16f);
  size_t co = (size_t)h * CHD + lane * 4;
  float4 vv = *(const float4*)(v + (size_t)s * DD + co);
  if (eproj) {
    union { unsigned long long u; __bf16 hx[4]; } ev;
    ev.u = *(const unsigned long long*)(eproj + (size_t)e * DD + co);
    vv.x += (float)ev.hx[0]; vv.y += (float)ev.hx[1];
    vv.z += (float)ev.hx[2]; vv.w += (float)ev.hx[3];
  }
  float* o = out + (size_t)d * DD + co;
  atomicAdd(o + 0, w * vv.x);
  atomicAdd(o + 1, w * vv.y);
  atomicAdd(o + 2, w * vv.z);
  atomicAdd(o + 3, w * vv.w);
}

// =====================================================================
// BatchNorm (training-mode batch stats, matches reference) + ReLU
// =====================================================================
__global__ __launch_bounds__(256) void bn_stats_kernel(
    const float* __restrict__ x, float* __restrict__ mu, float* __restrict__ rs,
    int Nrows) {
  int d = blockIdx.x;  // column
  float s = 0.f, s2 = 0.f;
  for (int r = threadIdx.x; r < Nrows; r += 256) {
    float v = x[(size_t)r * DD + d];
    s += v; s2 += v * v;
  }
  __shared__ float sh0[256], sh1[256];
  sh0[threadIdx.x] = s; sh1[threadIdx.x] = s2;
  __syncthreads();
  for (int st = 128; st; st >>= 1) {
    if (threadIdx.x < st) {
      sh0[threadIdx.x] += sh0[threadIdx.x + st];
      sh1[threadIdx.x] += sh1[threadIdx.x + st];
    }
    __syncthreads();
  }
  if (threadIdx.x == 0) {
    float m = sh0[0] / (float)Nrows;
    float var = sh1[0] / (float)Nrows - m * m;
    mu[d] = m;
    rs[d] = rsqrtf(var + 1e-5f);
  }
}

__global__ __launch_bounds__(256) void bn_relu_kernel(
    const float* __restrict__ x, float* __restrict__ y,
    const float* __restrict__ g, const float* __restrict__ b,
    const float* __restrict__ mu, const float* __restrict__ rs, size_t n) {
  size_t i = (size_t)blockIdx.x * 256 + threadIdx.x;
  if (i >= n) return;
  int d = (int)(i & (DD - 1));
  float v = g[d] * (x[i] - mu[d]) * rs[d] + b[d];
  y[i] = v > 0.f ? v : 0.f;
}

// =====================================================================
// pack / unpack / edge-gen glue
// =====================================================================
__global__ __launch_bounds__(256) void pack1_kernel(
    const float* __restrict__ text, const float* __restrict__ x,
    float* __restrict__ o) {
  size_t i = (size_t)blockIdx.x * 256 + threadIdx.x;
  if (i >= (size_t)N1C * DD) return;
  int n = (int)(i / DD), d = (int)(i % DD);
  int g = n / (NPG + 1), j = n % (NPG + 1);
  o[i] = (j == 0) ? text[(size_t)g * DD + d]
                  : x[((size_t)g * NPG + j - 1) * DD + d];
}

__global__ __launch_bounds__(256) void unpack1_kernel(
    const float* __restrict__ h1, float* __restrict__ o) {
  size_t i = (size_t)blockIdx.x * 256 + threadIdx.x;
  if (i >= (size_t)N2C * DD) return;
  int n = (int)(i / DD), d = (int)(i % DD);
  int g = n / NPG, j = n % NPG;
  o[i] = h1[((size_t)g * (NPG + 1) + 1 + j) * DD + d];
}

__global__ __launch_bounds__(256) void pack3_kernel(
    const float* __restrict__ gtok, const float* __restrict__ h,
    float* __restrict__ o) {
  size_t i = (size_t)blockIdx.x * 256 + threadIdx.x;
  if (i >= (size_t)N3C * DD) return;
  int n = (int)(i / DD), d = (int)(i % DD);
  int g = n / (NPG + TT), j = n % (NPG + TT);
  o[i] = (j < TT) ? gtok[(size_t)j * DD + d]
                  : h[((size_t)g * NPG + (j - TT)) * DD + d];
}

__global__ __launch_bounds__(256) void extract_out_kernel(
    const float* __restrict__ h3, float* __restrict__ o) {
  size_t i = (size_t)blockIdx.x * 256 + threadIdx.x;
  if (i >= (size_t)GN * TT * DD) return;
  int d = (int)(i % DD);
  int t = (int)((i / DD) % TT);
  int g = (int)(i / ((size_t)TT * DD));
  o[i] = h3[((size_t)g * (NPG + TT) + t) * DD + d];
}

__global__ __launch_bounds__(256) void gen_edges1_kernel(int* s, int* d) {
  int i = blockIdx.x * 256 + threadIdx.x;
  if (i >= E1C) return;
  int g = i / NPG, j = i % NPG;
  s[i] = g * (NPG + 1);
  d[i] = g * (NPG + 1) + 1 + j;
}

__global__ __launch_bounds__(256) void gen_edges3_kernel(int* s, int* d) {
  int i = blockIdx.x * 256 + threadIdx.x;
  if (i >= E3C) return;
  int g = i / (TT * NPG);
  int rem = i % (TT * NPG);
  int t = rem / NPG, n = rem % NPG;
  int base = g * (NPG + TT);
  d[i] = base + t;       // token (dst)
  s[i] = base + TT + n;  // node (src)
}

// =====================================================================
// host orchestration
// =====================================================================
extern "C" void kernel_launch(void* const* d_in, const int* in_sizes, int n_in,
                              void* d_out, int out_size, void* d_ws,
                              size_t ws_size, hipStream_t stream) {
  (void)in_sizes; (void)n_in; (void)out_size; (void)ws_size;

  // ---- input pointers (setup_inputs dict insertion order, pytree-flat) ----
  const float* x_in  = (const float*)d_in[0];
  const float* text  = (const float*)d_in[1];
  const float* eattr = (const float*)d_in[2];
  const int*   eidx  = (const int*)d_in[3];
  const float* mlpW1 = (const float*)d_in[4];
  const float* mlpB1 = (const float*)d_in[5];
  const float* mlpW2 = (const float*)d_in[6];
  const float* mlpB2 = (const float*)d_in[7];
  const float* gtok  = (const float*)d_in[8];

  struct ConvP { const float *Wq,*bq,*Wk,*bk,*Wv,*bv,*Ws,*bs,*We; };
  auto conv = [&](int base, bool edge) -> ConvP {
    ConvP p;
    p.Wq=(const float*)d_in[base+0]; p.bq=(const float*)d_in[base+1];
    p.Wk=(const float*)d_in[base+2]; p.bk=(const float*)d_in[base+3];
    p.Wv=(const float*)d_in[base+4]; p.bv=(const float*)d_in[base+5];
    p.Ws=(const float*)d_in[base+6]; p.bs=(const float*)d_in[base+7];
    p.We = edge ? (const float*)d_in[base+8] : nullptr;
    return p;
  };
  ConvP qn0 = conv(9,  false), qn1 = conv(17, false);
  const float* qn_g = (const float*)d_in[25]; const float* qn_b = (const float*)d_in[26];
  ConvP nn0 = conv(27, true),  nn1 = conv(36, true);
  const float* nn_g = (const float*)d_in[45]; const float* nn_b = (const float*)d_in[46];
  ConvP gt0 = conv(47, false), gt1 = conv(55, false);
  const float* gt_g = (const float*)d_in[63]; const float* gt_b = (const float*)d_in[64];

  // ---- workspace carve ----
  char* ws = (char*)d_ws;
  size_t off = 0;
  auto take = [&](size_t bytes) -> char* {
    char* p = ws + off;
    off += (bytes + 255) & ~(size_t)255;
    return p;
  };
  float*  bufA  = (float*)take((size_t)NMAXC * DD * 4);
  float*  bufB  = (float*)take((size_t)NMAXC * DD * 4);
  float*  bufC  = (float*)take((size_t)NMAXC * DD * 4);
  bf16_t* xb    = (bf16_t*)take((size_t)NMAXC * DD * 2);
  float*  qbuf  = (float*)take((size_t)NMAXC * DD * 4);
  float*  kbuf  = (float*)take((size_t)NMAXC * DD * 4);
  float*  vbuf  = (float*)take((size_t)NMAXC * DD * 4);
  bf16_t* ebf   = (bf16_t*)take((size_t)E2C * DD * 2);
  bf16_t* eproj = (bf16_t*)take((size_t)E2C * DD * 2);
  bf16_t* wtbuf = (bf16_t*)take((size_t)28 * DD * DD * 2);
  float*  alpha = (float*)take((size_t)E2C * HH * 4);
  unsigned* amax = (unsigned*)take((size_t)NMAXC * HH * 4);
  float*  den   = (float*)take((size_t)NMAXC * HH * 4);
  float*  mu    = (float*)take(DD * 4);
  float*  rs    = (float*)take(DD * 4);
  int*    src1  = (int*)take((size_t)E1C * 4);
  int*    dst1  = (int*)take((size_t)E1C * 4);
  int*    src3  = (int*)take((size_t)E3C * 4);
  int*    dst3  = (int*)take((size_t)E3C * 4);
  bf16_t* tb    = (bf16_t*)take((size_t)GN * DD * 2);
  float*  texth = (float*)take((size_t)GN * DD * 4);
  bf16_t* thb   = (bf16_t*)take((size_t)GN * DD * 2);
  float*  texto = (float*)take((size_t)GN * DD * 4);

  auto wt = [&](int slot) -> bf16_t* { return wtbuf + (size_t)slot * DD * DD; };

  // ---- transpose+cast all weight matrices to bf16 [N,K] ----
  // slots: 0 mlpW1, 1 mlpW2, 2-5 qn0(q,k,v,s), 6-9 qn1, 10-13 nn0, 14 nn0.We,
  //        15-18 nn1, 19 nn1.We, 20-23 gt0, 24-27 gt1
  const float* wsrc[28] = {
      mlpW1, mlpW2,
      qn0.Wq, qn0.Wk, qn0.Wv, qn0.Ws,
      qn1.Wq, qn1.Wk, qn1.Wv, qn1.Ws,
      nn0.Wq, nn0.Wk, nn0.Wv, nn0.Ws, nn0.We,
      nn1.Wq, nn1.Wk, nn1.Wv, nn1.Ws, nn1.We,
      gt0.Wq, gt0.Wk, gt0.Wv, gt0.Ws,
      gt1.Wq, gt1.Wk, gt1.Wv, gt1.Ws};
  {
    size_t n = (size_t)DD * DD;
    unsigned blocks = (unsigned)((n + 255) / 256);
    for (int i = 0; i < 28; ++i)
      transpose_cast_kernel<<<blocks, 256, 0, stream>>>(wsrc[i], wt(i), DD, DD);
  }

  auto castv = [&](const float* s, bf16_t* d, size_t n) {
    cast_f32_bf16_kernel<<<(unsigned)((n + 255) / 256), 256, 0, stream>>>(s, d, n);
  };
  auto gemm_f = [&](const bf16_t* A, const bf16_t* Bt, const float* bias,
                    float* C, int M, bool relu) {
    dim3 g((M + 127) / 128, DD / 128);
    if (relu)
      gemm_wmma<true, float><<<g, dim3(256), 0, stream>>>(A, Bt, bias, C, M, DD, DD);
    else
      gemm_wmma<false, float><<<g, dim3(256), 0, stream>>>(A, Bt, bias, C, M, DD, DD);
  };
  auto gemm_b = [&](const bf16_t* A, const bf16_t* Bt, bf16_t* C, int M) {
    dim3 g((M + 127) / 128, DD / 128);
    gemm_wmma<false, bf16_t><<<g, dim3(256), 0, stream>>>(A, Bt, nullptr, C, M, DD, DD);
  };

  const float scale = 0.08838834764831845f;  // 1/sqrt(128)

  auto run_tconv = [&](const float* xin, float* xout, int Nn, int slotQ,
                       const ConvP& p, const int* sp, const int* dp, int Ee,
                       const bf16_t* ep) {
    castv(xin, xb, (size_t)Nn * DD);
    gemm_f(xb, wt(slotQ + 0), p.bq, qbuf, Nn, false);
    gemm_f(xb, wt(slotQ + 1), p.bk, kbuf, Nn, false);
    gemm_f(xb, wt(slotQ + 2), p.bv, vbuf, Nn, false);
    gemm_f(xb, wt(slotQ + 3), p.bs, xout, Nn, false);  // skip path; attn adds in
    size_t nh = (size_t)Nn * HH;
    fill_u32_kernel<<<(unsigned)((nh + 255) / 256), 256, 0, stream>>>(amax, 0u, nh);
    fill_u32_kernel<<<(unsigned)((nh + 255) / 256), 256, 0, stream>>>((unsigned*)den, 0u, nh);
    int EH = Ee * HH;
    edge_alpha_kernel<<<(EH + 7) / 8, 256, 0, stream>>>(qbuf, kbuf, ep, sp, dp,
                                                        alpha, amax, EH, scale);
    edge_exp_kernel<<<(EH + 255) / 256, 256, 0, stream>>>(alpha, amax, den, dp, EH);
    edge_scatter_kernel<<<(EH + 7) / 8, 256, 0, stream>>>(alpha, den, vbuf, ep,
                                                          sp, dp, xout, EH);
  };
  auto run_bn = [&](const float* xi, float* yo, int Nn, const float* g,
                    const float* b) {
    bn_stats_kernel<<<DD, 256, 0, stream>>>(xi, mu, rs, Nn);
    size_t n = (size_t)Nn * DD;
    bn_relu_kernel<<<(unsigned)((n + 255) / 256), 256, 0, stream>>>(xi, yo, g, b, mu, rs, n);
  };

  // ---- query_node_mlp: Linear -> ReLU -> Linear on text embeddings ----
  castv(text, tb, (size_t)GN * DD);
  gemm_f(tb, wt(0), mlpB1, texth, GN, true);
  castv(texth, thb, (size_t)GN * DD);
  gemm_f(thb, wt(1), mlpB2, texto, GN, false);

  // ---- stage 1: question-node star graph ----
  gen_edges1_kernel<<<(E1C + 255) / 256, 256, 0, stream>>>(src1, dst1);
  {
    size_t n = (size_t)N1C * DD;
    pack1_kernel<<<(unsigned)((n + 255) / 256), 256, 0, stream>>>(texto, x_in, bufA);
  }
  run_tconv(bufA, bufB, N1C, 2, qn0, src1, dst1, E1C, nullptr);
  run_bn(bufB, bufC, N1C, qn_g, qn_b);
  run_tconv(bufC, bufB, N1C, 6, qn1, src1, dst1, E1C, nullptr);
  {
    size_t n = (size_t)N2C * DD;
    unpack1_kernel<<<(unsigned)((n + 255) / 256), 256, 0, stream>>>(bufB, bufA);
  }

  // ---- stage 2: node-node with edge features ----
  const int* src2 = eidx;
  const int* dst2 = eidx + E2C;
  castv(eattr, ebf, (size_t)E2C * DD);
  gemm_b(ebf, wt(14), eproj, E2C);                 // e @ We (conv 0)
  run_tconv(bufA, bufB, N2C, 10, nn0, src2, dst2, E2C, eproj);
  run_bn(bufB, bufC, N2C, nn_g, nn_b);
  gemm_b(ebf, wt(19), eproj, E2C);                 // e @ We (conv 1)
  run_tconv(bufC, bufB, N2C, 15, nn1, src2, dst2, E2C, eproj);

  // ---- stage 3: graph-token attention ----
  gen_edges3_kernel<<<(E3C + 255) / 256, 256, 0, stream>>>(src3, dst3);
  {
    size_t n = (size_t)N3C * DD;
    pack3_kernel<<<(unsigned)((n + 255) / 256), 256, 0, stream>>>(gtok, bufB, bufA);
  }
  run_tconv(bufA, bufB, N3C, 20, gt0, src3, dst3, E3C, nullptr);
  run_bn(bufB, bufC, N3C, gt_g, gt_b);
  run_tconv(bufC, bufB, N3C, 24, gt1, src3, dst3, E3C, nullptr);

  // ---- graph_token pooling: emit [G, T, D] ----
  {
    size_t n = (size_t)GN * TT * DD;
    extract_out_kernel<<<(unsigned)((n + 255) / 256), 256, 0, stream>>>(bufB, (float*)d_out);
  }
}
